// Sot_15668040695823
// MI455X (gfx1250) — matrix operand order repaired and checked by reading, not compiled, not run on hardware
//
#include <hip/hip_runtime.h>
#include <hip/hip_bf16.h>

// ---------------------------------------------------------------------------
// Sinkhorn-OT on cosine distances, N=8192, K=128, 20 iters (40 LSE sweeps).
// V kept in f16 (2 MB, L2-resident); S = V*V^T recomputed every sweep with
// v_wmma_f32_16x16x32_f16 instead of streaming a 256 MB log_k from HBM 41x.
// All exp arguments are provably in [-10.5, 0.5] -> no LSE max shift needed.
// B tiles are staged into LDS in 16 KB chunks, double-buffered with
// global_load_async_to_lds_b128 (ASYNCcnt) when the toolchain exposes it.
// Diagonal handled analytically: row sum accumulates unmasked, then
// exp(a[row]) (== the j==i term, since S_ii ~= 1) is subtracted once.
// ---------------------------------------------------------------------------

#define N_ROWS 8192
#define K_DIM  128
#define WAVES_PER_BLOCK 4                    // 128 threads / block
#define GRID_BLOCKS (N_ROWS / (16 * WAVES_PER_BLOCK))   // 128
#define NCHUNK 64                            // columns staged per round (16 KB)
#define NCHUNKS (N_ROWS / NCHUNK)            // 128

#if defined(__has_builtin)
#  if __has_builtin(__builtin_amdgcn_global_load_async_to_lds_b128)
#    define SOT_ASYNC 1
#  endif
#endif
#ifndef SOT_ASYNC
#  define SOT_ASYNC 0
#endif

typedef __attribute__((ext_vector_type(16))) _Float16 v16h;
typedef __attribute__((ext_vector_type(8)))  float    v8f;
typedef __attribute__((ext_vector_type(4)))  _Float16 v4h;
typedef __attribute__((ext_vector_type(4)))  int      v4i;

// Address-space-qualified v4i for the async-to-LDS builtin (AS1 = global,
// AS3 = LDS; clang's gfx1250 builtin takes typed v4i pointers).
typedef __attribute__((address_space(1))) v4i gv4i;
typedef __attribute__((address_space(3))) v4i lv4i;

union ABReg { v16h v; uint4 q[2]; };

// ---------------------------------------------------------------------------
// Kernel 1: row-normalize x (fp32) -> V (f16).  One wave per row.
// ---------------------------------------------------------------------------
__global__ void sot_normalize(const float* __restrict__ x,
                              _Float16* __restrict__ Vh) {
    const int wave = (blockIdx.x * blockDim.x + threadIdx.x) >> 5;
    const int lane = threadIdx.x & 31;
    const float4 xv = ((const float4*)(x + (size_t)wave * K_DIM))[lane];
    float ss = xv.x * xv.x + xv.y * xv.y + xv.z * xv.z + xv.w * xv.w;
    ss += __shfl_xor(ss, 1);
    ss += __shfl_xor(ss, 2);
    ss += __shfl_xor(ss, 4);
    ss += __shfl_xor(ss, 8);
    ss += __shfl_xor(ss, 16);
    const float sc = rsqrtf(ss);
    v4h hv;
    hv.x = (_Float16)(xv.x * sc);
    hv.y = (_Float16)(xv.y * sc);
    hv.z = (_Float16)(xv.z * sc);
    hv.w = (_Float16)(xv.w * sc);
    ((v4h*)(Vh + (size_t)wave * K_DIM))[lane] = hv;
}

// ---------------------------------------------------------------------------
// Kernel 2: zero the initial log_v vector.
// ---------------------------------------------------------------------------
__global__ void sot_zero(float* __restrict__ p) {
    p[blockIdx.x * blockDim.x + threadIdx.x] = 0.0f;
}

// ---------------------------------------------------------------------------
// Operand gathers (ISA 7.12.2 layouts):
//   A (16x32 f16): lane (h = lane/16, m = m0 + lane%16) holds halves
//     k0 + 8h + [0..7] and k0 + 16 + 8h + [0..7].
//   B (32x16 f16) with B = V^T: lane (n = lane%16) holds halves of V row n
//     k0 + 8h + [0..7] and k0 + 16 + 8h + [0..7].
// ---------------------------------------------------------------------------
__device__ __forceinline__ void load_A_tile(const _Float16* __restrict__ Vh,
                                            int m0, int lane, ABReg A[4]) {
    const int h = lane >> 4;
    const _Float16* arow = Vh + (size_t)(m0 + (lane & 15)) * K_DIM;
#pragma unroll
    for (int c = 0; c < 4; ++c) {
        A[c].q[0] = *(const uint4*)(arow + 32 * c + 8 * h);
        A[c].q[1] = *(const uint4*)(arow + 32 * c + 16 + 8 * h);
    }
}

__device__ __forceinline__ void load_B_tile(const _Float16* __restrict__ Bsrow0,
                                            int lane, ABReg B[4]) {
    const int h = lane >> 4;
    const _Float16* brow = Bsrow0 + (size_t)(lane & 15) * K_DIM;
#pragma unroll
    for (int c = 0; c < 4; ++c) {
        B[c].q[0] = *(const uint4*)(brow + 32 * c + 8 * h);
        B[c].q[1] = *(const uint4*)(brow + 32 * c + 16 + 8 * h);
    }
}

// ---------------------------------------------------------------------------
// Stage one 16 KB chunk (rows j0..j0+63 of V) into LDS.
// 128 threads x 128 B each.  Async path: 8x global_load_async_to_lds_b128
// per thread (ASYNCcnt); fallback: VGPR bounce (loads + ds_store_b128).
// ---------------------------------------------------------------------------
__device__ __forceinline__ void stage_issue(const _Float16* __restrict__ Vh,
                                            _Float16* __restrict__ dstBase,
                                            int j0, int t) {
    const char* src = (const char*)(Vh + (size_t)j0 * K_DIM) + (size_t)t * 128;
    char*       dst = (char*)dstBase + (size_t)t * 128;
#if SOT_ASYNC
#pragma unroll
    for (int i = 0; i < 8; ++i) {
        __builtin_amdgcn_global_load_async_to_lds_b128(
            (gv4i*)(src + 16 * i),
            (lv4i*)(dst + 16 * i),
            0, 0);
    }
#else
#pragma unroll
    for (int i = 0; i < 8; ++i)
        ((uint4*)dst)[i] = ((const uint4*)src)[i];
#endif
}

__device__ __forceinline__ void stage_wait() {
#if SOT_ASYNC
#  if __has_builtin(__builtin_amdgcn_s_wait_asynccnt)
    __builtin_amdgcn_s_wait_asynccnt(0);
#  else
    asm volatile("s_wait_asynccnt 0x0" ::: "memory");
#  endif
#endif
}

// ---------------------------------------------------------------------------
// Kernel 3: one Sinkhorn half-sweep.
//   bout[i] = -log( sum_{j != i} exp( 0.2*S[i][j] - 0.2 + a[j] ) )
// Each wave owns one 16-row M-tile; double-buffered 64-column chunks; 4
// chained WMMAs (K=128) per 16-col sub-tile; unmasked sum, diagonal term
// exp(a[i]) subtracted in the epilogue.
// ---------------------------------------------------------------------------
__global__ void __launch_bounds__(32 * WAVES_PER_BLOCK)
sot_lse_pass(const _Float16* __restrict__ Vh,
             const float* __restrict__ a,
             float* __restrict__ bout) {
    __shared__ __align__(16) _Float16 Bs[2][NCHUNK * K_DIM];

    const int t       = threadIdx.x;
    const int lane    = t & 31;
    const int wid     = t >> 5;
    const int h       = lane >> 4;
    const int m0      = (blockIdx.x * WAVES_PER_BLOCK + wid) * 16;
    const int rowbase = m0 + 8 * h;

    ABReg A[4];
    load_A_tile(Vh, m0, lane, A);

    float s[8];
#pragma unroll
    for (int e = 0; e < 8; ++e) s[e] = 0.0f;

    stage_issue(Vh, &Bs[0][0], 0, t);

    for (int ci = 0; ci < NCHUNKS; ++ci) {
        const int cur = ci & 1;
        stage_wait();                    // my async writes into Bs[cur] done
        __syncthreads();                 // everyone's are done
        if (ci + 1 < NCHUNKS)
            stage_issue(Vh, &Bs[cur ^ 1][0], (ci + 1) * NCHUNK, t);

        const int j0 = ci * NCHUNK;
#pragma unroll
        for (int sub = 0; sub < NCHUNK / 16; ++sub) {
            ABReg B[4];
            load_B_tile(&Bs[cur][(size_t)(16 * sub) * K_DIM], lane, B);

            v8f acc = {};
#pragma unroll
            for (int c = 0; c < 4; ++c) {
                acc = __builtin_amdgcn_wmma_f32_16x16x32_f16(
                    false, A[c].v, false, B[c].v, (short)0, acc, false, false);
            }

            const int   col = j0 + 16 * sub + (lane & 15);
            const float avc = a[col] - 0.2f;      // a[j] - 0.2 folded once
#pragma unroll
            for (int e = 0; e < 8; ++e)
                s[e] += __expf(0.2f * acc[e] + avc);
        }
    }

    // Reduce row sums across the 16-lane half (C layout: VGPR e <-> rows
    // m0+e for lanes 0-15, m0+8+e for lanes 16-31), subtract diagonal term.
#pragma unroll
    for (int e = 0; e < 8; ++e) {
        float se = s[e];
        se += __shfl_xor(se, 1);
        se += __shfl_xor(se, 2);
        se += __shfl_xor(se, 4);
        se += __shfl_xor(se, 8);
        if ((lane & 15) == 0) {
            const int row = rowbase + e;
            bout[row] = -__logf(se - __expf(a[row]));
        }
    }
}

// ---------------------------------------------------------------------------
// Kernel 4: finalize.  W[i][j] = exp(u[i] + 0.2*S - 0.2 + v[j]), diag = 1.
// Same GEMM structure; writes the 256 MB output (the only HBM-heavy step).
// ---------------------------------------------------------------------------
__global__ void __launch_bounds__(32 * WAVES_PER_BLOCK)
sot_finalize(const _Float16* __restrict__ Vh,
             const float* __restrict__ u,
             const float* __restrict__ v,
             float* __restrict__ W) {
    __shared__ __align__(16) _Float16 Bs[2][NCHUNK * K_DIM];

    const int t       = threadIdx.x;
    const int lane    = t & 31;
    const int wid     = t >> 5;
    const int h       = lane >> 4;
    const int m0      = (blockIdx.x * WAVES_PER_BLOCK + wid) * 16;
    const int rowbase = m0 + 8 * h;

    ABReg A[4];
    load_A_tile(Vh, m0, lane, A);

    float ue[8];
#pragma unroll
    for (int e = 0; e < 8; ++e) ue[e] = u[rowbase + e] - 0.2f;

    stage_issue(Vh, &Bs[0][0], 0, t);

    for (int ci = 0; ci < NCHUNKS; ++ci) {
        const int cur = ci & 1;
        stage_wait();
        __syncthreads();
        if (ci + 1 < NCHUNKS)
            stage_issue(Vh, &Bs[cur ^ 1][0], (ci + 1) * NCHUNK, t);

        const int j0 = ci * NCHUNK;
#pragma unroll
        for (int sub = 0; sub < NCHUNK / 16; ++sub) {
            ABReg B[4];
            load_B_tile(&Bs[cur][(size_t)(16 * sub) * K_DIM], lane, B);

            v8f acc = {};
#pragma unroll
            for (int c = 0; c < 4; ++c) {
                acc = __builtin_amdgcn_wmma_f32_16x16x32_f16(
                    false, A[c].v, false, B[c].v, (short)0, acc, false, false);
            }

            const int   col = j0 + 16 * sub + (lane & 15);
            const float vv  = v[col];
#pragma unroll
            for (int e = 0; e < 8; ++e) {
                float w = __expf(0.2f * acc[e] + (ue[e] + vv));
                if (col == rowbase + e) w = 1.0f;
                W[(size_t)(rowbase + e) * N_ROWS + col] = w;
            }
        }
    }
}

// ---------------------------------------------------------------------------
// Host launcher.  d_ws layout: [0, 2MB) f16 V ; then u (32 KB) ; v (32 KB).
// ---------------------------------------------------------------------------
extern "C" void kernel_launch(void* const* d_in, const int* in_sizes, int n_in,
                              void* d_out, int out_size, void* d_ws, size_t ws_size,
                              hipStream_t stream) {
    (void)in_sizes; (void)n_in; (void)out_size; (void)ws_size;

    const float* x  = (const float*)d_in[0];
    float*       W  = (float*)d_out;
    _Float16*    Vh = (_Float16*)d_ws;
    float*       u  = (float*)((char*)d_ws + (size_t)N_ROWS * K_DIM * sizeof(_Float16));
    float*       v  = u + N_ROWS;

    // V = normalize_rows(x) in f16 (2 MB, stays L2-resident for all passes).
    sot_normalize<<<N_ROWS / 8, 256, 0, stream>>>(x, Vh);
    // log_v0 = 0
    sot_zero<<<N_ROWS / 256, 256, 0, stream>>>(v);

    // 20 Sinkhorn iterations; log_k is symmetric, so both updates are the
    // same row-LSE with the roles of u and v swapped.
    for (int it = 0; it < 20; ++it) {
        sot_lse_pass<<<GRID_BLOCKS, 32 * WAVES_PER_BLOCK, 0, stream>>>(Vh, v, u);
        sot_lse_pass<<<GRID_BLOCKS, 32 * WAVES_PER_BLOCK, 0, stream>>>(Vh, u, v);
    }

    sot_finalize<<<GRID_BLOCKS, 32 * WAVES_PER_BLOCK, 0, stream>>>(Vh, u, v, W);
}
